// TransformerBlock_74294344286691
// MI455X (gfx1250) — matrix operand order repaired and checked by reading, not compiled, not run on hardware
//
#include <hip/hip_runtime.h>
#include <math.h>

typedef _Float16 half_t;
typedef _Float16 h8  __attribute__((ext_vector_type(8)));
typedef _Float16 v16h __attribute__((ext_vector_type(16)));
typedef float    v8f  __attribute__((ext_vector_type(8)));

#define DEVINL __device__ __forceinline__

// D = A(16x32 f16) x B(32x16 f16) + C(16x16 f32)
DEVINL v8f wmma16(v16h a, v16h b, v8f c) {
  return __builtin_amdgcn_wmma_f32_16x16x32_f16(false, a, false, b, (short)0, c,
                                                false, false);
}

// Async DMA: 16B per lane, global -> LDS, tracked by ASYNCcnt.
// VDST vgpr = LDS byte address (low 32 bits of flat shared addr), addr = SADDR + VOFF.
DEVINL void async_ld_b128(const void* lds_dst, const void* gbase, int goff) {
  unsigned ldsa = (unsigned)(size_t)lds_dst;
  unsigned long long gb = (unsigned long long)(size_t)gbase;
  asm volatile("global_load_async_to_lds_b128 %0, %1, %2"
               :
               : "v"(ldsa), "v"(goff), "s"(gb)
               : "memory");
}
DEVINL void async_wait0() {
  asm volatile("s_wait_asynccnt 0x0" ::: "memory");
}

// A-tile (16x32, f16): lane<16 -> row=lane, K in {0..7, 16..23}; lane>=16 -> K in {8..15, 24..31}
DEVINL v16h load_a_tile(const half_t* p, int row0, int stride, int k0, int lane) {
  int row = row0 + (lane & 15);
  int kb  = (lane >> 4) << 3;
  const half_t* q = p + row * stride + k0 + kb;
  union { v16h v; h8 h[2]; } u;
  u.h[0] = *(const h8*)(q);
  u.h[1] = *(const h8*)(q + 16);
  return u.v;
}

// B-tile (32x16, f16) with B[k][n] = src[n0+n][k0+k]:
// lane<16 -> col=lane, K in {0..7, 16..23}; lane>=16 -> K in {8..15, 24..31}.
DEVINL v16h load_bt_tile(const half_t* p, int n0, int stride, int k0, int lane) {
  int n  = n0 + (lane & 15);
  int kb = (lane >> 4) << 3;
  const half_t* q = p + n * stride + k0 + kb;
  union { v16h v; h8 h[2]; } u;
  u.h[0] = *(const h8*)(q);
  u.h[1] = *(const h8*)(q + 16);
  return u.v;
}

// ---------------- Kernel 0: relative-position-bias MLP -> [3][64][64] f32 ----
__global__ __launch_bounds__(256) void k_bias_mlp(
    const float* __restrict__ m1w, const float* __restrict__ m1b,
    const float* __restrict__ m2w, const float* __restrict__ m2b,
    float* __restrict__ bias_out) {
  int tid = blockIdx.x * 256 + threadIdx.x;
  if (tid >= 4096) return;
  int i = tid >> 6, j = tid & 63;
  int di = (i >> 3) - (j >> 3);
  int dj = (i & 7) - (j & 7);
  float f0 = (di == 0) ? 0.f : (di > 0 ? log1pf((float)di) : -log1pf((float)(-di)));
  float f1 = (dj == 0) ? 0.f : (dj > 0 ? log1pf((float)dj) : -log1pf((float)(-dj)));
  float a0 = 0.f, a1 = 0.f, a2 = 0.f;
  for (int k = 0; k < 256; ++k) {
    float h = f0 * m1w[2 * k] + f1 * m1w[2 * k + 1] + m1b[k];
    h = fmaxf(h, 0.f);
    a0 += h * m2w[k];
    a1 += h * m2w[256 + k];
    a2 += h * m2w[512 + k];
  }
  bias_out[tid]        = a0 + m2b[0];
  bias_out[4096 + tid] = a1 + m2b[1];
  bias_out[8192 + tid] = a2 + m2b[2];
}

// ---------------- Kernel 1: LayerNorm + repack x,prompt to window-major f16 --
__global__ __launch_bounds__(64) void k_ln_pack(
    const float* __restrict__ x, const float* __restrict__ prompt,
    const float* __restrict__ ln_g, const float* __restrict__ ln_b,
    half_t* __restrict__ xn_w, half_t* __restrict__ pr_w) {
  __shared__ float  s_f[64 * 96];
  __shared__ half_t s_h[64 * 96];
  int win = blockIdx.x, t = threadIdx.x;
  int b = win >> 10, wr = (win >> 5) & 31, wc = win & 31;
  int h = wr * 8 + (t >> 3), w = wc * 8 + (t & 7);
  size_t pbase = (size_t)b * 96 * 65536 + (size_t)h * 256 + w;
  for (int c = 0; c < 96; ++c) s_f[t * 96 + c] = x[pbase + (size_t)c * 65536];
  float mu = 0.f, s2 = 0.f;
  for (int c = 0; c < 96; ++c) { float v = s_f[t * 96 + c]; mu += v; s2 += v * v; }
  mu *= (1.f / 96.f);
  float var  = s2 * (1.f / 96.f) - mu * mu;
  float rstd = rsqrtf(var + 1e-5f);
  for (int c = 0; c < 96; ++c)
    s_h[t * 96 + c] = (half_t)(((s_f[t * 96 + c] - mu) * rstd) * ln_g[c] + ln_b[c]);
  __syncthreads();
  {
    const uint4* src = (const uint4*)s_h;
    uint4* dst = (uint4*)(xn_w + (size_t)win * 6144);
    for (int i = t; i < 768; i += 64) dst[i] = src[i];
  }
  __syncthreads();
  for (int c = 0; c < 96; ++c)
    s_h[t * 96 + c] = (half_t)prompt[pbase + (size_t)c * 65536];
  __syncthreads();
  {
    const uint4* src = (const uint4*)s_h;
    uint4* dst = (uint4*)(pr_w + (size_t)win * 6144);
    for (int i = t; i < 768; i += 64) dst[i] = src[i];
  }
}

// ---------------- Kernel 2: QKV projection (WMMA GEMM per window) ------------
__global__ __launch_bounds__(256) void k_qkv(
    const half_t* __restrict__ xn_w, const half_t* __restrict__ pr_w,
    const float* __restrict__ wq, const float* __restrict__ wk,
    const float* __restrict__ wv, const float* __restrict__ bq,
    const float* __restrict__ bk, const float* __restrict__ bv,
    half_t* __restrict__ q_w, half_t* __restrict__ k_w, half_t* __restrict__ v_w) {
  __shared__ half_t s_a[64 * 96];
  __shared__ half_t s_w[144 * 96];
  int win = blockIdx.x, t = threadIdx.x;
  int lane = t & 31, wid = t >> 5;
  int l15 = lane & 15, hi8 = (lane >> 4) << 3;
  size_t base = (size_t)win * 6144;
  // async stage of the xn window tile (12KB contiguous)
  for (int i = t; i < 768; i += 256)
    async_ld_b128((const char*)s_a + i * 16, xn_w + base, i * 16);
  __builtin_prefetch(xn_w + base + 6144, 0, 0);  // next window activation tile
  async_wait0();
  for (int ph = 0; ph < 2; ++ph) {
    __syncthreads();
    for (int i = t; i < 13824; i += 256) {
      int rowl = i / 96, cin = i - rowl * 96;
      int grow = ph * 144 + rowl;
      int sect = grow / 96, o = grow - sect * 96;
      const float* wsel = (sect == 0) ? wq : (sect == 1 ? wk : wv);
      s_w[i] = (half_t)wsel[o * 96 + cin];
    }
    __syncthreads();
    for (int ti = wid; ti < 36; ti += 8) {
      int mt = ti / 9, ntl = ti - (ti / 9) * 9;
      int nt = ph * 9 + ntl;
      v8f acc = {0.f, 0.f, 0.f, 0.f, 0.f, 0.f, 0.f, 0.f};
#pragma unroll
      for (int ks = 0; ks < 96; ks += 32) {
        v16h a  = load_a_tile(s_a, mt * 16, 96, ks, lane);
        v16h bt = load_bt_tile(s_w, ntl * 16, 96, ks, lane);
        acc = wmma16(a, bt, acc);
      }
      int sect = nt / 6;
      int c = nt * 16 - sect * 96 + l15;
      const float* bsel = (sect == 0) ? bq : (sect == 1 ? bk : bv);
      float bb = bsel[c];
      half_t* dst = (sect == 0) ? q_w : (sect == 1 ? k_w : v_w);
#pragma unroll
      for (int r = 0; r < 8; ++r) {
        int n = mt * 16 + r + hi8;
        size_t idx = base + (size_t)n * 96 + c;
        float val = acc[r] + bb;
        if (sect == 0) val = (val + (float)pr_w[idx]) * 0.17677669529663687f;
        dst[idx] = (half_t)val;
      }
    }
  }
}

// ---------------- Kernel 3: windowed attention (WMMA) ------------------------
__global__ __launch_bounds__(96) void k_attn(
    const half_t* __restrict__ q_w, const half_t* __restrict__ k_w,
    const half_t* __restrict__ v_w, const float* __restrict__ bias,
    half_t* __restrict__ att) {
  __shared__ half_t s_q[64 * 96];
  __shared__ half_t s_k[64 * 96];
  __shared__ half_t s_vraw[64 * 96];
  __shared__ half_t s_vt[96 * 64];
  __shared__ half_t s_p[3 * 16 * 64];
  int win = blockIdx.x, t = threadIdx.x;
  size_t base = (size_t)win * 6144;
  // async stage q, k, v window tiles (3 x 12KB contiguous)
  for (int i = t; i < 768; i += 96) {
    async_ld_b128((const char*)s_q + i * 16, q_w + base, i * 16);
    async_ld_b128((const char*)s_k + i * 16, k_w + base, i * 16);
    async_ld_b128((const char*)s_vraw + i * 16, v_w + base, i * 16);
  }
  __builtin_prefetch(q_w + base + 6144, 0, 0);
  async_wait0();
  __syncthreads();
  // transpose v (LDS -> LDS) so P@V B-tiles are contiguous per lane
  for (int i = t; i < 3072; i += 96) {
    union { unsigned int u; half_t h[2]; } cv;
    cv.u = ((const unsigned int*)s_vraw)[i];
    int e = i * 2; int n = e / 96, c = e - n * 96;
    s_vt[c * 64 + n]       = cv.h[0];
    s_vt[(c + 1) * 64 + n] = cv.h[1];
  }
  __syncthreads();
  int head = t >> 5, lane = t & 31;
  int l15 = lane & 15, hi8 = (lane >> 4) << 3;
  const float* bh = bias + head * 4096;
  half_t* pl = s_p + head * 1024;
  for (int mt = 0; mt < 4; ++mt) {
    v16h a = load_a_tile(s_q, mt * 16, 96, head * 32, lane);
    v8f s[4];
#pragma unroll
    for (int nt = 0; nt < 4; ++nt) {
#pragma unroll
      for (int r = 0; r < 8; ++r)
        s[nt][r] = bh[(mt * 16 + r + hi8) * 64 + nt * 16 + l15];
      v16h bt = load_bt_tile(s_k, nt * 16, 96, head * 32, lane);
      s[nt] = wmma16(a, bt, s[nt]);
    }
#pragma unroll
    for (int r = 0; r < 8; ++r) {
      float m = fmaxf(fmaxf(s[0][r], s[1][r]), fmaxf(s[2][r], s[3][r]));
#pragma unroll
      for (int off = 8; off > 0; off >>= 1) m = fmaxf(m, __shfl_xor(m, off, 16));
      float sum = 0.f;
#pragma unroll
      for (int nt = 0; nt < 4; ++nt) {
        float e = __expf(s[nt][r] - m);
        s[nt][r] = e;
        sum += e;
      }
#pragma unroll
      for (int off = 8; off > 0; off >>= 1) sum += __shfl_xor(sum, off, 16);
      float inv = 1.f / sum;
#pragma unroll
      for (int nt = 0; nt < 4; ++nt)
        pl[(r + hi8) * 64 + nt * 16 + l15] = (half_t)(s[nt][r] * inv);
    }
    v8f o0 = {0.f, 0.f, 0.f, 0.f, 0.f, 0.f, 0.f, 0.f};
    v8f o1 = {0.f, 0.f, 0.f, 0.f, 0.f, 0.f, 0.f, 0.f};
#pragma unroll
    for (int kt = 0; kt < 2; ++kt) {
      v16h pa = load_a_tile(pl, 0, 64, kt * 32, lane);
      v16h b0 = load_bt_tile(s_vt, head * 32, 64, kt * 32, lane);
      o0 = wmma16(pa, b0, o0);
      v16h b1 = load_bt_tile(s_vt, head * 32 + 16, 64, kt * 32, lane);
      o1 = wmma16(pa, b1, o1);
    }
#pragma unroll
    for (int r = 0; r < 8; ++r) {
      int n = mt * 16 + r + hi8;
      size_t ob = base + (size_t)n * 96 + head * 32;
      att[ob + l15]      = (half_t)o0[r];
      att[ob + 16 + l15] = (half_t)o1[r];
    }
  }
}

// ---------------- Kernel 4: depthwise 5x5 reflect conv on V + attn add -------
__global__ __launch_bounds__(64) void k_conv(
    const half_t* __restrict__ v_w, const half_t* __restrict__ att,
    const float* __restrict__ wdw, const float* __restrict__ bdw,
    half_t* __restrict__ y_w) {
  __shared__ int    s_gidx[144];
  __shared__ half_t s_v[144 * 96];
  __shared__ float  s_wd[96 * 25];
  int win = blockIdx.x, t = threadIdx.x;
  int b = win >> 10, wr = (win >> 5) & 31, wc = win & 31;
  int h0 = wr * 8, w0 = wc * 8;
  for (int i = t; i < 144; i += 64) {
    int hi = i / 12, wi = i - (i / 12) * 12;
    int ph = h0 + hi - 2; ph = ph < 0 ? -ph : (ph > 255 ? 510 - ph : ph);
    int pw = w0 + wi - 2; pw = pw < 0 ? -pw : (pw > 255 ? 510 - pw : pw);
    s_gidx[i] = ((b * 1024 + (ph >> 3) * 32 + (pw >> 3)) << 6) + ((ph & 7) << 3) + (pw & 7);
  }
  for (int i = t; i < 2400; i += 64) s_wd[i] = wdw[i];
  __syncthreads();
  // async halo gather: per-lane 16B chunks from reflected window positions
  for (int i = t; i < 1728; i += 64) {
    int pos = i / 12, ch = (i - (i / 12) * 12) * 8;
    async_ld_b128((const char*)s_v + pos * 192 + ch * 2, v_w,
                  s_gidx[pos] * 192 + ch * 2);
  }
  async_wait0();
  __syncthreads();
  int r = t >> 3, cl = t & 7;
  size_t obase = (size_t)win * 6144 + (size_t)t * 96;
  for (int c = 0; c < 96; ++c) {
    float acc = bdw[c];
#pragma unroll
    for (int ky = 0; ky < 5; ++ky)
#pragma unroll
      for (int kx = 0; kx < 5; ++kx)
        acc += (float)s_v[((r + ky) * 12 + (cl + kx)) * 96 + c] * s_wd[c * 25 + ky * 5 + kx];
    y_w[obase + c] = (half_t)(acc + (float)att[obase + c]);
  }
}

// ---------------- Kernel 5: output projection (WMMA) + residual -> NCHW ------
__global__ __launch_bounds__(256) void k_proj(
    const half_t* __restrict__ y_w, const float* __restrict__ wp,
    const float* __restrict__ bp, const float* __restrict__ x,
    float* __restrict__ out) {
  __shared__ half_t s_y[64 * 96];
  __shared__ half_t s_w[96 * 96];
  int win = blockIdx.x, t = threadIdx.x;
  int lane = t & 31, wid = t >> 5;
  int l15 = lane & 15, hi8 = (lane >> 4) << 3;
  int b = win >> 10, wr = (win >> 5) & 31, wc = win & 31;
  size_t base = (size_t)win * 6144;
  for (int i = t; i < 768; i += 256)
    async_ld_b128((const char*)s_y + i * 16, y_w + base, i * 16);
  for (int i = t; i < 9216; i += 256) s_w[i] = (half_t)wp[i];
  __builtin_prefetch(y_w + base + 6144, 0, 0);
  async_wait0();
  __syncthreads();
  for (int ti = wid; ti < 24; ti += 8) {
    int mt = ti >> 2, nt = ti & 3;
    v8f acc = {0.f, 0.f, 0.f, 0.f, 0.f, 0.f, 0.f, 0.f};
#pragma unroll
    for (int ks = 0; ks < 96; ks += 32) {
      v16h a  = load_a_tile(s_w, mt * 16, 96, ks, lane);
      v16h bt = load_bt_tile(s_y, nt * 16, 96, ks, lane);
      acc = wmma16(a, bt, acc);
    }
    int n = nt * 16 + l15;
    int h = wr * 8 + (n >> 3), w = wc * 8 + (n & 7);
#pragma unroll
    for (int r = 0; r < 8; ++r) {
      int cout = mt * 16 + r + hi8;
      size_t idx = (((size_t)b * 96 + cout) * 256 + h) * 256 + w;
      out[idx] = x[idx] + acc[r] + bp[cout];
    }
  }
}

// -----------------------------------------------------------------------------
extern "C" void kernel_launch(void* const* d_in, const int* in_sizes, int n_in,
                              void* d_out, int out_size, void* d_ws, size_t ws_size,
                              hipStream_t stream) {
  (void)in_sizes; (void)n_in; (void)out_size; (void)ws_size;
  const float* x      = (const float*)d_in[0];
  const float* prompt = (const float*)d_in[1];
  const float* ln_g   = (const float*)d_in[2];
  const float* ln_b   = (const float*)d_in[3];
  const float* wq  = (const float*)d_in[4];
  const float* bq  = (const float*)d_in[5];
  const float* wk  = (const float*)d_in[6];
  const float* bk  = (const float*)d_in[7];
  const float* wv  = (const float*)d_in[8];
  const float* bv  = (const float*)d_in[9];
  const float* wp  = (const float*)d_in[10];
  const float* bp  = (const float*)d_in[11];
  const float* wdw = (const float*)d_in[12];
  const float* bdw = (const float*)d_in[13];
  const float* m1w = (const float*)d_in[14];
  const float* m1b = (const float*)d_in[15];
  const float* m2w = (const float*)d_in[16];
  const float* m2b = (const float*)d_in[17];
  float* out = (float*)d_out;

  char* ws = (char*)d_ws;
  float* bias = (float*)ws;                       // 3*64*64 f32 = 48 KB
  const size_t HB = (size_t)4096 * 6144;          // halfs per [Npix][96] buffer
  half_t* xn  = (half_t*)(ws + 65536);
  half_t* pr  = xn + HB;
  half_t* qb  = pr + HB;
  half_t* kb  = qb + HB;
  half_t* vb  = kb + HB;
  half_t* attb = pr;  // prompt dead after k_qkv -> reuse for attention output
  half_t* yb   = xn;  // xn dead after k_qkv   -> reuse for conv+attn sum

  k_bias_mlp<<<16, 256, 0, stream>>>(m1w, m1b, m2w, m2b, bias);
  k_ln_pack<<<4096, 64, 0, stream>>>(x, prompt, ln_g, ln_b, xn, pr);
  k_qkv<<<4096, 256, 0, stream>>>(xn, pr, wq, wk, wv, bq, bk, bv, qb, kb, vb);
  k_attn<<<4096, 96, 0, stream>>>(qb, kb, vb, bias, attb);
  k_conv<<<4096, 64, 0, stream>>>(vb, attb, wdw, bdw, yb);
  k_proj<<<4096, 256, 0, stream>>>(yb, wp, bp, x, out);
}
